// energy_latencyGNN_4_1_41446434406431
// MI455X (gfx1250) — compile-verified, matrix operand order checked by reading, not run on hardware
//
#include <hip/hip_runtime.h>

// CDNA5 / gfx1250 fused GNN + MLP kernel, round 2.
// Change vs round 1: register-blocked WMMA loop. A-fragments for the wave's
// two 16-row M tiles are preloaded once per layer (64 VGPRs), and each
// B-fragment LDS load now feeds two back-to-back v_wmma_f32_16x16x32_f16
// (one per M tile). This cuts ds_load_b128 traffic ~3x and lengthens the
// WMMA->WMMA dependency distance (fewer hazard v_nops).

typedef __attribute__((ext_vector_type(16))) _Float16 v16h;
typedef __attribute__((ext_vector_type(8)))  float    v8f;
typedef __attribute__((ext_vector_type(8)))  short    short8;

#define LRELU_SLOPE 0.01f

struct KArgs { const void* in[36]; float* out; };

// ---------------- LDS layout (bytes) ----------------
// w1t : 128x128 f16  @      0  (32768)
// w2t : 128x128 f16  @  32768  (32768)
// w3t :  64x128 f16  @  65536  (16384)
// w4t :  16x 64 f16  @  81920  ( 2048)
// b1  : 128 f32      @  83968  (  512)
// b2  : 128 f32      @  84480  (  512)
// b3  :  64 f32      @  84992  (  256)
// b4  :  16 f32      @  85248  (   64)
// adjF:  64 f32      @  85312  (  256)
// adjB:  64 f32      @  85568  (  256)
// buf0: 256x128 f16  @  85824  (65536)
// buf1: 256x128 f16  @ 151360  (65536)
// total = 216896 bytes (< 320 KB per workgroup on CDNA5)
#define SMEM_TOTAL 216896

__device__ __forceinline__ float sigm_(float x)  { return 1.0f / (1.0f + __expf(-x)); }
__device__ __forceinline__ float lrelu_(float x) { return fmaxf(x, 0.0f) + LRELU_SLOPE * fminf(x, 0.0f); }

// Load a 16-half A/B fragment for v_wmma_f32_16x16x32_f16.
// CDNA5 16-bit fragment layout: lanes 0-15 hold K = kOff..kOff+7 and
// kOff+16..kOff+23; lanes 16-31 the +8 shifted ranges (caller bakes the
// lane-half shift into kOff). Two ds_load_b128 per fragment.
__device__ __forceinline__ v16h frag16(const _Float16* base, int row, int pitch, int kOff) {
  union { v16h v; short8 s[2]; } u;
  const _Float16* p = base + row * pitch + kOff;
  u.s[0] = *(const short8*)(p);
  u.s[1] = *(const short8*)(p + 16);
  return u.v;
}

// One dense layer on WMMA: out = leakyrelu(A @ W^T + bias).
// A   : [256 x KPAD] f16 in LDS (row-major, KPAD multiple of 32)
// WT  : [NTILES*16 x KPAD] f16 in LDS (output-channel-major)
// outB: [256 x npad] f16 in LDS
// Each wave owns two 16-row M tiles; A-fragments are register-resident,
// each B-fragment load feeds both M tiles.
template<int KPAD, int NTILES>
__device__ __forceinline__ void mlp_layer(const _Float16* A, const _Float16* WT,
                                          const float* bias, _Float16* outB,
                                          int npad, int wave, int lane) {
  const int halfSel = (lane & 16) ? 8 : 0;   // K shift for src frags, M shift for acc
  const int lm = lane & 15;
  constexpr int KB = KPAD / 32;

  v16h afr[2][KB];
  #pragma unroll
  for (int t = 0; t < 2; ++t) {
    const int rowA = (wave * 2 + t) * 16 + lm;
    #pragma unroll
    for (int kb = 0; kb < KB; ++kb)
      afr[t][kb] = frag16(A, rowA, KPAD, kb * 32 + halfSel);
  }

  for (int nt = 0; nt < NTILES; ++nt) {
    const int nIdx = nt * 16 + lm;
    v8f acc0 = {}, acc1 = {};
    #pragma unroll
    for (int kb = 0; kb < KB; ++kb) {
      v16h bf = frag16(WT, nIdx, KPAD, kb * 32 + halfSel);
      acc0 = __builtin_amdgcn_wmma_f32_16x16x32_f16(false, afr[0][kb], false, bf,
                                                    (short)0, acc0, false, false);
      acc1 = __builtin_amdgcn_wmma_f32_16x16x32_f16(false, afr[1][kb], false, bf,
                                                    (short)0, acc1, false, false);
    }
    const float bn = bias[nIdx];
    #pragma unroll
    for (int j = 0; j < 8; ++j) {
      // C/D layout: VGPR j -> M=j for lanes 0-15, M=j+8 for lanes 16-31
      const int r0 = (wave * 2 + 0) * 16 + j + halfSel;
      const int r1 = (wave * 2 + 1) * 16 + j + halfSel;
      outB[r0 * npad + nIdx] = (_Float16)lrelu_(acc0[j] + bn);
      outB[r1 * npad + nIdx] = (_Float16)lrelu_(acc1[j] + bn);
    }
  }
}

// Final layer: [256x64] @ W4^T(16x64, cols>=2 zero) + b4, sigmoid, store 2 cols.
__device__ __forceinline__ void mlp_final(const _Float16* A, const _Float16* WT,
                                          const float* bias, float* out,
                                          int blockBase, int wave, int lane) {
  const int halfSel = (lane & 16) ? 8 : 0;
  const int lm = lane & 15;

  v16h afr[2][2];
  #pragma unroll
  for (int t = 0; t < 2; ++t) {
    const int rowA = (wave * 2 + t) * 16 + lm;
    #pragma unroll
    for (int kb = 0; kb < 2; ++kb)
      afr[t][kb] = frag16(A, rowA, 64, kb * 32 + halfSel);
  }

  v8f acc0 = {}, acc1 = {};
  #pragma unroll
  for (int kb = 0; kb < 2; ++kb) {
    v16h bf = frag16(WT, lm, 64, kb * 32 + halfSel);
    acc0 = __builtin_amdgcn_wmma_f32_16x16x32_f16(false, afr[0][kb], false, bf,
                                                  (short)0, acc0, false, false);
    acc1 = __builtin_amdgcn_wmma_f32_16x16x32_f16(false, afr[1][kb], false, bf,
                                                  (short)0, acc1, false, false);
  }
  const float bn = bias[lm];
  if (lm < 2) {
    #pragma unroll
    for (int j = 0; j < 8; ++j) {
      const int r0 = (wave * 2 + 0) * 16 + j + halfSel;
      const int r1 = (wave * 2 + 1) * 16 + j + halfSel;
      out[(size_t)(blockBase + r0) * 2 + lm] = sigm_(acc0[j] + bn);
      out[(size_t)(blockBase + r1) * 2 + lm] = sigm_(acc1[j] + bn);
    }
  }
}

// 3 gated graph layers for one direction; h0 = node scalars, adj = 8x8 counts.
// P[0..3]=layer0 (W(1,5),b(5),gw(1,1),gb(1)); P[4..7],P[8..11]=layers 1,2 (fi=5).
// Writes 40 f16 features (node-major, emb-minor) to feat.
__device__ __forceinline__ void gnn_chain(const float h0[8], const float* adj,
                                          const void* const* P, _Float16* feat) {
  float h[8][5];
  {
    const float* W0  = (const float*)P[0];
    const float* bb0 = (const float*)P[1];
    const float  gw0 = ((const float*)P[2])[0];
    const float  gb0 = ((const float*)P[3])[0];
    float gated[8];
    #pragma unroll
    for (int i = 0; i < 8; ++i) gated[i] = h0[i] * sigm_(h0[i] * gw0 + gb0);
    float agg[8];
    #pragma unroll
    for (int i = 0; i < 8; ++i) {
      float s = 0.0f;
      #pragma unroll
      for (int n = 0; n < 8; ++n) s += adj[i * 8 + n] * gated[n];
      agg[i] = s;
    }
    #pragma unroll
    for (int i = 0; i < 8; ++i)
      #pragma unroll
      for (int j = 0; j < 5; ++j)
        h[i][j] = lrelu_(agg[i] * W0[j] + bb0[j]);
  }
  #pragma unroll
  for (int L = 1; L < 3; ++L) {
    const float* W  = (const float*)P[L * 4 + 0];   // (5,5) row-major
    const float* bb = (const float*)P[L * 4 + 1];
    const float* gw = (const float*)P[L * 4 + 2];   // (5,1)
    const float  gb = ((const float*)P[L * 4 + 3])[0];
    float gated[8][5];
    #pragma unroll
    for (int i = 0; i < 8; ++i) {
      float z = gb;
      #pragma unroll
      for (int j = 0; j < 5; ++j) z += h[i][j] * gw[j];
      const float g = sigm_(z);
      #pragma unroll
      for (int j = 0; j < 5; ++j) gated[i][j] = h[i][j] * g;
    }
    float agg[8][5];
    #pragma unroll
    for (int i = 0; i < 8; ++i)
      #pragma unroll
      for (int j = 0; j < 5; ++j) {
        float s = 0.0f;
        #pragma unroll
        for (int n = 0; n < 8; ++n) s += adj[i * 8 + n] * gated[n][j];
        agg[i][j] = s;
      }
    float hn[8][5];
    #pragma unroll
    for (int i = 0; i < 8; ++i)
      #pragma unroll
      for (int j2 = 0; j2 < 5; ++j2) {
        float s = bb[j2];
        #pragma unroll
        for (int j = 0; j < 5; ++j) s += agg[i][j] * W[j * 5 + j2];
        hn[i][j2] = lrelu_(s);
      }
    #pragma unroll
    for (int i = 0; i < 8; ++i)
      #pragma unroll
      for (int j = 0; j < 5; ++j) h[i][j] = hn[i][j];
  }
  #pragma unroll
  for (int i = 0; i < 8; ++i)
    #pragma unroll
    for (int j = 0; j < 5; ++j) feat[i * 5 + j] = (_Float16)h[i][j];
}

__global__ __launch_bounds__(256) void gnn_mlp_kernel(KArgs a) {
  extern __shared__ char smem[];
  _Float16* w1t  = (_Float16*)(smem + 0);
  _Float16* w2t  = (_Float16*)(smem + 32768);
  _Float16* w3t  = (_Float16*)(smem + 65536);
  _Float16* w4t  = (_Float16*)(smem + 81920);
  float*    b1   = (float*)(smem + 83968);
  float*    b2   = (float*)(smem + 84480);
  float*    b3   = (float*)(smem + 84992);
  float*    b4   = (float*)(smem + 85248);
  float*    adjF = (float*)(smem + 85312);
  float*    adjB = (float*)(smem + 85568);
  _Float16* buf0 = (_Float16*)(smem + 85824);    // feat / ping
  _Float16* buf1 = (_Float16*)(smem + 151360);   // pong

  const int tid = threadIdx.x;

  // ---- Phase 0: stage f16 transposed weights + biases + adjacency in LDS ----
  {
    const float* W1 = (const float*)a.in[28];   // (104,128)
    for (int idx = tid; idx < 128 * 128; idx += 256) {
      const int n = idx >> 7, k = idx & 127;
      const float v = (k < 104) ? W1[k * 128 + n] : 0.0f;
      w1t[n * 128 + k] = (_Float16)v;
    }
    const float* W2 = (const float*)a.in[30];   // (128,128)
    for (int idx = tid; idx < 128 * 128; idx += 256) {
      const int n = idx >> 7, k = idx & 127;
      w2t[n * 128 + k] = (_Float16)W2[k * 128 + n];
    }
    const float* W3 = (const float*)a.in[32];   // (128,64)
    for (int idx = tid; idx < 64 * 128; idx += 256) {
      const int n = idx >> 7, k = idx & 127;
      w3t[n * 128 + k] = (_Float16)W3[k * 64 + n];
    }
    const float* W4 = (const float*)a.in[34];   // (64,2) -> pad N to 16
    for (int idx = tid; idx < 16 * 64; idx += 256) {
      const int n = idx >> 6, k = idx & 63;
      const float v = (n < 2) ? W4[k * 2 + n] : 0.0f;
      w4t[n * 64 + k] = (_Float16)v;
    }
    const float* mb1 = (const float*)a.in[29];
    const float* mb2 = (const float*)a.in[31];
    const float* mb3 = (const float*)a.in[33];
    const float* mb4 = (const float*)a.in[35];
    if (tid < 128) { b1[tid] = mb1[tid]; b2[tid] = mb2[tid]; }
    if (tid < 64)  { b3[tid] = mb3[tid]; }
    if (tid < 16)  { b4[tid] = (tid < 2) ? mb4[tid] : 0.0f; }
    if (tid == 0) {
      const int* src = (const int*)a.in[2];
      const int* dst = (const int*)a.in[3];
      for (int i = 0; i < 64; ++i) { adjF[i] = 0.0f; adjB[i] = 0.0f; }
      for (int e = 0; e < 16; ++e) {
        const int s = src[e], t = dst[e];
        adjF[t * 8 + s] += 1.0f;   // forward: agg[dst] += gated[src]
        adjB[s * 8 + t] += 1.0f;   // reverse: agg[src] += gated[dst]
      }
    }
  }
  __syncthreads();

  // ---- Phase A: per-row GNN (scalar VALU), emit f16 feature rows ----
  {
    const int b = blockIdx.x * 256 + tid;
    const float* data = (const float*)a.in[0];
    float h0[8];
    #pragma unroll
    for (int i = 0; i < 8; ++i) h0[i] = data[(size_t)b * 8 + i];

    _Float16* feat = buf0 + tid * 128;
    gnn_chain(h0, adjF, a.in + 4,  feat);        // x  -> feat[0..39]
    gnn_chain(h0, adjB, a.in + 16, feat + 40);   // xx -> feat[40..79]
    const float* dm = (const float*)a.in[1];
    #pragma unroll
    for (int m = 0; m < 24; ++m) feat[80 + m] = (_Float16)dm[(size_t)b * 24 + m];
    #pragma unroll
    for (int k = 104; k < 128; ++k) feat[k] = (_Float16)0.0f;
  }
  __syncthreads();

  // ---- Phase B: WMMA MLP, ping-pong through LDS ----
  const int wave = tid >> 5, lane = tid & 31;
  mlp_layer<128, 8>(buf0, w1t, b1, buf1, 128, wave, lane);   // 104(->128) -> 128
  __syncthreads();
  mlp_layer<128, 8>(buf1, w2t, b2, buf0, 128, wave, lane);   // 128 -> 128
  __syncthreads();
  mlp_layer<128, 4>(buf0, w3t, b3, buf1, 64, wave, lane);    // 128 -> 64
  __syncthreads();
  mlp_final(buf1, w4t, b4, a.out, blockIdx.x * 256, wave, lane); // 64 -> 2, sigmoid
}

extern "C" void kernel_launch(void* const* d_in, const int* in_sizes, int n_in,
                              void* d_out, int out_size, void* d_ws, size_t ws_size,
                              hipStream_t stream) {
  (void)d_ws; (void)ws_size; (void)out_size;
  KArgs a;
  for (int i = 0; i < 36; ++i) a.in[i] = (i < n_in) ? d_in[i] : nullptr;
  a.out = (float*)d_out;

  const int B = in_sizes[0] / 8;          // data is (B, 8, 1)
  const int nblocks = B / 256;            // 262144 / 256 = 1024

  // Allow >64KB dynamic LDS (CDNA5 workgroups may use up to 320KB).
  static bool attrSet = false;            // host-side config, not stream work
  if (!attrSet) {
    hipFuncSetAttribute((const void*)gnn_mlp_kernel,
                        hipFuncAttributeMaxDynamicSharedMemorySize, SMEM_TOTAL);
    attrSet = true;
  }
  hipLaunchKernelGGL(gnn_mlp_kernel, dim3(nblocks), dim3(256), SMEM_TOTAL, stream, a);
}
